// TransformerPointerGeneratorLM_15908558864807
// MI455X (gfx1250) — matrix operand order repaired and checked by reading, not compiled, not run on hardware
//
#include <hip/hip_runtime.h>
#include <hip/hip_bf16.h>

// ---------------------------------------------------------------------------
// Model constants (reference: B=2, S=1024, D=512, V=32000, H=8, L=4, DH=64)
// ---------------------------------------------------------------------------
#define CB 2
#define CS 1024
#define CD 512
#define CV 32000
#define CH 8
#define CL 4
#define CDH 64

typedef __attribute__((ext_vector_type(16))) __bf16 bf16x16;
typedef __attribute__((ext_vector_type(8)))  __bf16 bf16x8;
typedef __attribute__((ext_vector_type(8)))  float  f32x8;

// fp32 -> bf16 with round-to-nearest-even
__device__ __forceinline__ __bf16 f2bf(float f) {
  union { float f; unsigned u; } v; v.f = f;
  unsigned r = v.u + 0x7FFFu + ((v.u >> 16) & 1u);
  unsigned short s = (unsigned short)(r >> 16);
  return __builtin_bit_cast(__bf16, s);
}

__device__ __forceinline__ float block_reduce_sum256(float v, float* sbuf) {
  int tid = threadIdx.x;
  sbuf[tid] = v; __syncthreads();
  for (int s = 128; s > 0; s >>= 1) {
    if (tid < s) sbuf[tid] += sbuf[tid + s];
    __syncthreads();
  }
  float r = sbuf[0]; __syncthreads();
  return r;
}

__device__ __forceinline__ float block_reduce_max256(float v, float* sbuf) {
  int tid = threadIdx.x;
  sbuf[tid] = v; __syncthreads();
  for (int s = 128; s > 0; s >>= 1) {
    if (tid < s) sbuf[tid] = fmaxf(sbuf[tid], sbuf[tid + s]);
    __syncthreads();
  }
  float r = sbuf[0]; __syncthreads();
  return r;
}

// ---------------------------------------------------------------------------
// Batched WMMA GEMM:  C = alpha*(A x op(B)) + bias + res
//   Block tile 64(M) x 64(N), K-step 32, 256 threads = 8 waves (4x2).
//   Each wave: one 16x32 A-frag x two B-frags -> 2x v_wmma_f32_16x16x32_bf16.
//   LDS double-buffered; tiles staged in fragment-ready order so fragment
//   loads are 2x ds_load_b128 per fragment (no scalar u16 chains).
//   REQUIRES: M%64==0, N%64==0, K%32==0, lda/ldb/ldc%4==0 (true for all
//   GEMMs in this model) -- no bounds checks, no divergent staging.
//   Batch offsets:  off = (z/bdiv)*sXo + (z%bdiv)*sXi   (supports per-(b,h))
// ---------------------------------------------------------------------------
constexpr int GLW = 40;   // LDS row pitch in bf16 (80B: 16B-aligned, conflict-free)

template<bool TRANSB>
__global__ __launch_bounds__(256)
void k_gemm_wmma(const float* __restrict__ A, const float* __restrict__ Bg,
                 const float* __restrict__ bias, const float* __restrict__ res,
                 float* __restrict__ C,
                 int M, int N, int K, int lda, int ldb, int ldc,
                 int bdiv,
                 long sAo, long sAi, long sBo, long sBi, long sCo, long sCi,
                 float alpha)
{
  __shared__ __align__(16) __bf16 As[2][64 * GLW];
  __shared__ __align__(16) __bf16 Bs[2][64 * GLW];

  const int tid  = threadIdx.x;
  const int lane = tid & 31;
  const int wave = tid >> 5;
  const int wm   = wave & 3;    // 16-row tile 0..3
  const int wn   = wave >> 2;   // 32-col tile 0..1

  const int z = blockIdx.z;
  const float* Ab = A  + (long)(z / bdiv) * sAo + (long)(z % bdiv) * sAi;
  const float* Bb = Bg + (long)(z / bdiv) * sBo + (long)(z % bdiv) * sBi;
  const long offC = (long)(z / bdiv) * sCo + (long)(z % bdiv) * sCi;

  const int row0 = blockIdx.x * 64;
  const int col0 = blockIdx.y * 64;

  // ---- staging thread mappings -------------------------------------------
  // A: thread -> row (0..63), 8-wide k-group; store in natural k order
  // so a lane's 16 fragment elements (k = 16h + s) are contiguous in LDS.
  const int amr = tid >> 2;              // 0..63
  const int akg = (tid & 3) * 8;         // 0,8,16,24
  const int ag  = akg >> 3;
  const int apg = ag;                    // group
  const int aofs = amr * GLW + apg * 8;

  auto stageA = [&](int k0, int buf) {
    const float4* src = (const float4*)(Ab + (long)(row0 + amr) * lda + k0 + akg);
    float4 x0 = src[0];
    float4 x1 = src[1];
    bf16x8 p;
    p[0] = f2bf(x0.x); p[1] = f2bf(x0.y); p[2] = f2bf(x0.z); p[3] = f2bf(x0.w);
    p[4] = f2bf(x1.x); p[5] = f2bf(x1.y); p[6] = f2bf(x1.z); p[7] = f2bf(x1.w);
    *(bf16x8*)&As[buf][aofs] = p;
  };

  // B: store tile as [n][k] (k contiguous) so fragment loads are wide.
  const int btn  = tid >> 2;             // TRANSB: n 0..63
  const int btkg = (tid & 3) * 8;        // TRANSB: k-group
  const int bnk  = tid >> 3;             // !TRANSB: k 0..31
  const int bnn8 = (tid & 7) * 8;        // !TRANSB: n-group (8 wide)

  auto stageB = [&](int k0, int buf) {
    if (TRANSB) {
      const float4* src = (const float4*)(Bb + (long)(col0 + btn) * ldb + k0 + btkg);
      float4 x0 = src[0];
      float4 x1 = src[1];
      bf16x8 p;
      p[0] = f2bf(x0.x); p[1] = f2bf(x0.y); p[2] = f2bf(x0.z); p[3] = f2bf(x0.w);
      p[4] = f2bf(x1.x); p[5] = f2bf(x1.y); p[6] = f2bf(x1.z); p[7] = f2bf(x1.w);
      *(bf16x8*)&Bs[buf][btn * GLW + btkg] = p;
    } else {
      const float4* src = (const float4*)(Bb + (long)(k0 + bnk) * ldb + col0 + bnn8);
      float4 x0 = src[0];
      float4 x1 = src[1];
      __bf16 vals[8] = { f2bf(x0.x), f2bf(x0.y), f2bf(x0.z), f2bf(x0.w),
                         f2bf(x1.x), f2bf(x1.y), f2bf(x1.z), f2bf(x1.w) };
      #pragma unroll
      for (int j = 0; j < 8; ++j) Bs[buf][(bnn8 + j) * GLW + bnk] = vals[j];
    }
  };

  // ---- fragment read offsets (contiguous 16 bf16 per lane) ---------------
  const int arow  = (wm * 16 + (lane & 15)) * GLW + (lane >> 4) * 16;
  const int b0row = (wn * 32 + (lane & 15)) * GLW + (lane >> 4) * 16;
  const int b1row = b0row + 16 * GLW;

  f32x8 acc0 = {}, acc1 = {};

  stageA(0, 0);
  stageB(0, 0);
  int buf = 0;
  for (int k0 = 0; k0 < K; k0 += 32) {
    __syncthreads();
    if (k0 + 32 < K) { stageA(k0 + 32, buf ^ 1); stageB(k0 + 32, buf ^ 1); }

    bf16x8 alo = *(const bf16x8*)&As[buf][arow];
    bf16x8 ahi = *(const bf16x8*)&As[buf][arow + 8];
    bf16x16 af = __builtin_shufflevector(alo, ahi,
        0, 1, 2, 3, 4, 5, 6, 7, 8, 9, 10, 11, 12, 13, 14, 15);

    bf16x8 b0l = *(const bf16x8*)&Bs[buf][b0row];
    bf16x8 b0h = *(const bf16x8*)&Bs[buf][b0row + 8];
    bf16x16 bf0 = __builtin_shufflevector(b0l, b0h,
        0, 1, 2, 3, 4, 5, 6, 7, 8, 9, 10, 11, 12, 13, 14, 15);

    bf16x8 b1l = *(const bf16x8*)&Bs[buf][b1row];
    bf16x8 b1h = *(const bf16x8*)&Bs[buf][b1row + 8];
    bf16x16 bf1 = __builtin_shufflevector(b1l, b1h,
        0, 1, 2, 3, 4, 5, 6, 7, 8, 9, 10, 11, 12, 13, 14, 15);

    acc0 = __builtin_amdgcn_wmma_f32_16x16x32_bf16(false, af, false, bf0,
                                                   (short)0, acc0, false, false);
    acc1 = __builtin_amdgcn_wmma_f32_16x16x32_bf16(false, af, false, bf1,
                                                   (short)0, acc1, false, false);
    buf ^= 1;
  }

  // ---- epilogue: C/D layout n = lane&15, m = 8*(lane>>4)+j ---------------
  float* Cb = C + offC;
  const float* Rb = res ? (res + offC) : nullptr;
  const int gr0 = row0 + wm * 16 + (lane >> 4) * 8;
  const int gc0 = col0 + wn * 32 + (lane & 15);
  #pragma unroll
  for (int j = 0; j < 8; ++j) {
    long idx = (long)(gr0 + j) * ldc + gc0;
    float v0 = alpha * acc0[j];
    float v1 = alpha * acc1[j];
    if (bias) { v0 += bias[gc0]; v1 += bias[gc0 + 16]; }
    if (Rb)   { v0 += Rb[idx];   v1 += Rb[idx + 16]; }
    Cb[idx]      = v0;
    Cb[idx + 16] = v1;
  }
}

// ---------------------------------------------------------------------------
// Elementwise / reduction kernels
// ---------------------------------------------------------------------------
__global__ void k_embed(const int* __restrict__ x, const float* __restrict__ tok,
                        float* __restrict__ emb, long n) {
  long i = (long)blockIdx.x * blockDim.x + threadIdx.x;
  if (i >= n) return;
  long r = i >> 9;            // /D (D=512)
  int  c = (int)(i & 511);
  emb[i] = tok[(long)x[r] * CD + c];
}

__global__ __launch_bounds__(512)
void k_gru(const float* __restrict__ gx, const float* __restrict__ whh,
           const float* __restrict__ bhh, float* __restrict__ hout) {
  __shared__ __align__(16) float hsh[CD];
  const int i = threadIdx.x;           // hidden index 0..511
  const int b = blockIdx.x;
  hsh[i] = 0.0f;
  __syncthreads();
  const float* wr = whh + (long)i * CD;
  const float* wz = whh + (long)(CD + i) * CD;
  const float* wn = whh + (long)(2 * CD + i) * CD;
  const float br = bhh[i], bz = bhh[CD + i], bn = bhh[2 * CD + i];
  for (int t = 0; t < CS; ++t) {
    float gr = br, gz = bz, gn = bn;
    for (int d = 0; d < CD; d += 4) {
      float4 hv = *(const float4*)&hsh[d];
      float4 a  = *(const float4*)&wr[d];
      float4 c  = *(const float4*)&wz[d];
      float4 e  = *(const float4*)&wn[d];
      gr += a.x*hv.x + a.y*hv.y + a.z*hv.z + a.w*hv.w;
      gz += c.x*hv.x + c.y*hv.y + c.z*hv.z + c.w*hv.w;
      gn += e.x*hv.x + e.y*hv.y + e.z*hv.z + e.w*hv.w;
    }
    const float* gxt = gx + ((long)b * CS + t) * (3 * CD);
    float r  = 1.0f / (1.0f + __expf(-(gxt[i] + gr)));
    float zz = 1.0f / (1.0f + __expf(-(gxt[CD + i] + gz)));
    float nn = tanhf(gxt[2 * CD + i] + r * gn);
    float hn = (1.0f - zz) * nn + zz * hsh[i];
    __syncthreads();
    hsh[i] = hn;
    hout[((long)b * CS + t) * CD + i] = hn;
    __syncthreads();
  }
}

__global__ __launch_bounds__(256)
void k_layernorm(const float* __restrict__ in, const float* __restrict__ g,
                 const float* __restrict__ b, float* __restrict__ out) {
  __shared__ float sbuf[256];
  const long row = blockIdx.x;
  const float* x = in + row * CD;
  float s = 0.0f;
  for (int i = threadIdx.x; i < CD; i += 256) s += x[i];
  float mu = block_reduce_sum256(s, sbuf) * (1.0f / CD);
  float v = 0.0f;
  for (int i = threadIdx.x; i < CD; i += 256) { float d = x[i] - mu; v += d * d; }
  float var = block_reduce_sum256(v, sbuf) * (1.0f / CD);
  float inv = rsqrtf(var + 1e-5f);
  float* o = out + row * CD;
  for (int i = threadIdx.x; i < CD; i += 256) o[i] = (x[i] - mu) * inv * g[i] + b[i];
}

__global__ void k_rope(float* __restrict__ t, long npairs) {
  long idx = (long)blockIdx.x * blockDim.x + threadIdx.x;
  if (idx >= npairs) return;
  int  q  = (int)(idx & 255);      // pair within row (D/2 = 256)
  int  p  = q & 31;                // pair within head (DH/2 = 32)
  int  hh = q >> 5;                // head
  long bs = idx >> 8;              // b*S + s
  int  s  = (int)(bs & (CS - 1));
  float inv = __powf(10000.0f, -(float)p * (1.0f / 32.0f));
  float ang = (float)s * inv;
  float c = __cosf(ang), sn = __sinf(ang);
  long base = bs * CD + (long)hh * CDH;
  float t1 = t[base + p], t2 = t[base + 32 + p];
  t[base + p]      = t1 * c - t2 * sn;
  t[base + 32 + p] = t1 * sn + t2 * c;
}

// causal softmax over rows of length S, valid cols j<=i, pre-scale by `scale`
__global__ __launch_bounds__(256)
void k_causal_softmax(float* __restrict__ s, float scale) {
  __shared__ float sbuf[256];
  const int i = blockIdx.x;
  const int z = blockIdx.y;
  float* row = s + ((long)z * CS + i) * CS;
  const int n = i + 1;
  float m = -1e30f;
  for (int j = threadIdx.x; j < n; j += 256) m = fmaxf(m, row[j] * scale);
  m = block_reduce_max256(m, sbuf);
  float sum = 0.0f;
  for (int j = threadIdx.x; j < n; j += 256) sum += __expf(row[j] * scale - m);
  sum = block_reduce_sum256(sum, sbuf);
  float inv = 1.0f / sum;
  for (int j = threadIdx.x; j < CS; j += 256)
    row[j] = (j < n) ? __expf(row[j] * scale - m) * inv : 0.0f;
}

__global__ void k_gelu(float* __restrict__ x, long n) {
  long i = (long)blockIdx.x * blockDim.x + threadIdx.x;
  if (i >= n) return;
  float v = x[i];
  float c = 0.7978845608028654f * (v + 0.044715f * v * v * v);
  x[i] = 0.5f * v * (1.0f + tanhf(c));
}

__global__ void k_concat(const float* __restrict__ t, const float* __restrict__ ctx,
                         float* __restrict__ comb, long total) {
  long i = (long)blockIdx.x * blockDim.x + threadIdx.x;
  if (i >= total) return;
  long r = i >> 10;  int c = (int)(i & 1023);
  comb[i] = (c < CD) ? t[r * CD + c] : ctx[r * CD + (c - CD)];
}

__global__ __launch_bounds__(256)
void k_gate(const float* __restrict__ comb, const float* __restrict__ gw,
            const float* __restrict__ gb, float* __restrict__ gate) {
  __shared__ float sbuf[256];
  long r = blockIdx.x;
  const float* c = comb + r * (2 * CD);
  float s = 0.0f;
  for (int i = threadIdx.x; i < 2 * CD; i += 256) s += c[i] * gw[i];
  s = block_reduce_sum256(s, sbuf);
  if (threadIdx.x == 0) gate[r] = 1.0f / (1.0f + __expf(-(s + gb[0])));
}

// out_row = gate * softmax(out_row) over V
__global__ __launch_bounds__(256)
void k_softmax_gate(float* __restrict__ out, const float* __restrict__ gate) {
  __shared__ float sbuf[256];
  long r = blockIdx.x;
  float* row = out + r * (long)CV;
  float m = -1e30f;
  for (int j = threadIdx.x; j < CV; j += 256) m = fmaxf(m, row[j]);
  m = block_reduce_max256(m, sbuf);
  float s = 0.0f;
  for (int j = threadIdx.x; j < CV; j += 256) s += __expf(row[j] - m);
  s = block_reduce_sum256(s, sbuf);
  float inv = gate[r] / s;
  for (int j = threadIdx.x; j < CV; j += 256) row[j] = __expf(row[j] - m) * inv;
}

// out[b,i,x[b,j]] += (1-gate[b,i]) * probs[b,i,j]   for j<=i
__global__ __launch_bounds__(256)
void k_scatter(float* __restrict__ out, const float* __restrict__ probs,
               const float* __restrict__ gate, const int* __restrict__ x) {
  int i = blockIdx.x;
  int b = blockIdx.y;
  float og = 1.0f - gate[(long)b * CS + i];
  const float* pr = probs + ((long)b * CS + i) * CS;
  float* orow = out + ((long)b * CS + i) * (long)CV;
  for (int j = threadIdx.x; j <= i; j += 256) {
    atomicAdd(&orow[x[(long)b * CS + j]], og * pr[j]);
  }
}

// ---------------------------------------------------------------------------
// Host-side GEMM wrapper
// ---------------------------------------------------------------------------
static void gemm(hipStream_t st, bool transb,
                 const float* A, const float* Bm, const float* bias,
                 const float* res, float* C,
                 int M, int N, int K, int lda, int ldb, int ldc,
                 int batch, int bdiv,
                 long sAo, long sAi, long sBo, long sBi, long sCo, long sCi,
                 float alpha) {
  dim3 grid((M + 63) / 64, (N + 63) / 64, batch);
  if (transb)
    k_gemm_wmma<true ><<<grid, 256, 0, st>>>(A, Bm, bias, res, C, M, N, K,
        lda, ldb, ldc, bdiv, sAo, sAi, sBo, sBi, sCo, sCi, alpha);
  else
    k_gemm_wmma<false><<<grid, 256, 0, st>>>(A, Bm, bias, res, C, M, N, K,
        lda, ldb, ldc, bdiv, sAo, sAi, sBo, sBi, sCo, sCi, alpha);
}

// ---------------------------------------------------------------------------
// kernel_launch
// input flattening order (setup_inputs dict order, params nested in order):
//  0:x 1:token_embed 2:abs_pos_embed 3:gru_w_ih 4:gru_w_hh 5:gru_b_ih
//  6:gru_b_hh  7..70: 4 blocks x {ln1_g,ln1_b,ln2_g,ln2_b,wq,bq,wk,bk,
//  wv,bv,wo,bo,w1,b1,w2,b2}  71:gen_w 72:gen_b 73:gate_w 74:gate_b
// ---------------------------------------------------------------------------
extern "C" void kernel_launch(void* const* d_in, const int* in_sizes, int n_in,
                              void* d_out, int out_size, void* d_ws, size_t ws_size,
                              hipStream_t stream) {
  (void)in_sizes; (void)n_in; (void)out_size; (void)ws_size;
  auto F = [&](int i) { return (const float*)d_in[i]; };
  const int*   x    = (const int*)d_in[0];
  const float* tok  = F(1);
  const float* pos  = F(2);
  const float* wih  = F(3);
  const float* whh  = F(4);
  const float* bih  = F(5);
  const float* bhh  = F(6);
  const float* genw = F(71);
  const float* genb = F(72);
  const float* gtw  = F(73);
  const float* gtb  = F(74);
  float* out = (float*)d_out;

  const int BS = CB * CS;            // 2048 rows
  const long M1 = (long)BS * CD;     // 1M floats

  float* ws    = (float*)d_ws;
  float* h_    = ws;                 // [B,S,D]          1M
  float* gxb   = h_   + M1;          // [B,S,3D]         3M
  float* abuf  = gxb  + 3 * M1;      // ln out / emb     1M
  float* qb    = abuf + M1;          // 1M
  float* kb    = qb   + M1;          // 1M
  float* vb    = kb   + M1;          // 1M
  float* ob    = vb   + M1;          // 1M
  float* sc    = ob   + M1;          // [B,H,S,S]        16M
  float* m1b   = sc   + 16 * M1;     // [B,S,4D]         4M
  float* ptrb  = m1b  + 4 * M1;      // [B,S,S]          2M
  float* ctx   = ptrb + 2 * M1;      // [B,S,D]          1M
  float* comb  = ctx  + M1;          // [B,S,2D]         2M
  float* gateb = comb + 2 * M1;      // [B*S]

  // 1. embedding
  { long n = M1; k_embed<<<(int)((n + 255) / 256), 256, 0, stream>>>(x, tok, abuf, n); }

  // 2. gx = emb @ w_ih^T + b_ih   (NT: w_ih is [3D,D])
  gemm(stream, true, abuf, wih, bih, nullptr, gxb,
       BS, 3 * CD, CD, CD, CD, 3 * CD, 1, 1, 0, 0, 0, 0, 0, 0, 1.0f);

  // 3. GRU scan -> h
  k_gru<<<CB, 512, 0, stream>>>(gxb, whh, bhh, h_);

  // 4. transformer blocks
  for (int l = 0; l < CL; ++l) {
    const int p0 = 7 + 16 * l;
    const float *ln1g = F(p0 + 0), *ln1b = F(p0 + 1);
    const float *ln2g = F(p0 + 2), *ln2b = F(p0 + 3);
    const float *wq = F(p0 + 4), *bq = F(p0 + 5);
    const float *wk = F(p0 + 6), *bk = F(p0 + 7);
    const float *wv = F(p0 + 8), *bv = F(p0 + 9);
    const float *wo = F(p0 + 10), *bo = F(p0 + 11);
    const float *w1 = F(p0 + 12), *b1 = F(p0 + 13);
    const float *w2 = F(p0 + 14), *b2 = F(p0 + 15);

    k_layernorm<<<BS, 256, 0, stream>>>(h_, ln1g, ln1b, abuf);

    gemm(stream, false, abuf, wq, bq, nullptr, qb, BS, CD, CD, CD, CD, CD,
         1, 1, 0, 0, 0, 0, 0, 0, 1.0f);
    gemm(stream, false, abuf, wk, bk, nullptr, kb, BS, CD, CD, CD, CD, CD,
         1, 1, 0, 0, 0, 0, 0, 0, 1.0f);
    gemm(stream, false, abuf, wv, bv, nullptr, vb, BS, CD, CD, CD, CD, CD,
         1, 1, 0, 0, 0, 0, 0, 0, 1.0f);

    { long n = M1 / 2;
      k_rope<<<(int)((n + 255) / 256), 256, 0, stream>>>(qb, n);
      k_rope<<<(int)((n + 255) / 256), 256, 0, stream>>>(kb, n); }

    // scores[b,h] = q_head @ k_head^T   (batch z = b*H+h)
    gemm(stream, true, qb, kb, nullptr, nullptr, sc,
         CS, CS, CDH, CD, CD, CS, CB * CH, CH,
         (long)CS * CD, CDH, (long)CS * CD, CDH,
         (long)CH * CS * CS, (long)CS * CS, 1.0f);

    { dim3 g(CS, CB * CH); k_causal_softmax<<<g, 256, 0, stream>>>(sc, 0.125f); }

    // o[b,h] = att @ v_head  (writes [B,S,D] layout directly via ldc=D)
    gemm(stream, false, sc, vb, nullptr, nullptr, ob,
         CS, CDH, CS, CS, CD, CD, CB * CH, CH,
         (long)CH * CS * CS, (long)CS * CS,
         (long)CS * CD, CDH, (long)CS * CD, CDH, 1.0f);

    // h = h + o @ wo + bo
    gemm(stream, false, ob, wo, bo, h_, h_, BS, CD, CD, CD, CD, CD,
         1, 1, 0, 0, 0, 0, 0, 0, 1.0f);

    // MLP
    k_layernorm<<<BS, 256, 0, stream>>>(h_, ln2g, ln2b, abuf);
    gemm(stream, false, abuf, w1, b1, nullptr, m1b, BS, 4 * CD, CD,
         CD, 4 * CD, 4 * CD, 1, 1, 0, 0, 0, 0, 0, 0, 1.0f);
    { long n = 4 * M1; k_gelu<<<(int)((n + 255) / 256), 256, 0, stream>>>(m1b, n); }
    gemm(stream, false, m1b, w2, b2, h_, h_, BS, CD, 4 * CD,
         4 * CD, CD, CD, 1, 1, 0, 0, 0, 0, 0, 0, 1.0f);
  }

  // 5. pointer scores: ptr = t t^T + t pos^T  (then scaled-causal softmax)
  gemm(stream, true, h_, h_, nullptr, nullptr, ptrb,
       CS, CS, CD, CD, CD, CS, CB, 1,
       (long)CS * CD, 0, (long)CS * CD, 0, (long)CS * CS, 0, 1.0f);
  gemm(stream, true, h_, pos, nullptr, ptrb, ptrb,
       CS, CS, CD, CD, CD, CS, CB, 1,
       (long)CS * CD, 0, 0, 0, (long)CS * CS, 0, 1.0f);
  { dim3 g(CS, CB);
    k_causal_softmax<<<g, 256, 0, stream>>>(ptrb, 0.04419417382415922f); } // 1/sqrt(512)

  // 6. context = probs @ t
  gemm(stream, false, ptrb, h_, nullptr, nullptr, ctx,
       CS, CD, CS, CS, CD, CD, CB, 1,
       (long)CS * CS, 0, (long)CS * CD, 0, (long)CS * CD, 0, 1.0f);

  // 7. comb = [t, context]
  { long n = (long)BS * 2 * CD;
    k_concat<<<(int)((n + 255) / 256), 256, 0, stream>>>(h_, ctx, comb, n); }

  // 8. logits -> d_out ; gate ; out = gate*softmax(logits)
  gemm(stream, false, comb, genw, genb, nullptr, out, BS, CV, 2 * CD,
       2 * CD, CV, CV, 1, 1, 0, 0, 0, 0, 0, 0, 1.0f);
  k_gate<<<BS, 256, 0, stream>>>(comb, gtw, gtb, gateb);
  k_softmax_gate<<<BS, 256, 0, stream>>>(out, gateb);

  // 9. out += (1-gate) * P_copy  (scatter, avoids 256MB P_copy buffer)
  { dim3 g(CS, CB); k_scatter<<<g, 256, 0, stream>>>(out, ptrb, gateb, x); }
}
